// TemporalSelfAttention_781684048572
// MI455X (gfx1250) — matrix operand order repaired and checked by reading, not compiled, not run on hardware
//
#include <hip/hip_runtime.h>
#include <hip/hip_bf16.h>
#include <math.h>

typedef _Float16 half_t;
typedef __attribute__((ext_vector_type(16))) _Float16 v16h;
typedef __attribute__((ext_vector_type(8)))  float    v8f;

union V16H { v16h v; uint4 u[2]; half_t h[16]; };

constexpr int kB  = 8;
constexpr int kT  = 512;
constexpr int kD  = 1024;
constexpr int kH  = 16;
constexpr int kHD = 64;
constexpr int kTE = 1024;

// ---------------- gfx1250 async global->LDS support (guarded; safe fallback)
#if defined(__has_builtin)
#if __has_builtin(__builtin_amdgcn_global_load_async_to_lds_b128)
#define ASYNC_LDS 1
#endif
#endif
#ifndef ASYNC_LDS
#define ASYNC_LDS 0
#endif

#if ASYNC_LDS
typedef int v4i_t __attribute__((vector_size(16)));
typedef __attribute__((address_space(1))) v4i_t* as1_v4i;
typedef __attribute__((address_space(3))) v4i_t* as3_v4i;

__device__ __forceinline__ void async_cp16(half_t* lds, const half_t* g) {
  __builtin_amdgcn_global_load_async_to_lds_b128(
      (as1_v4i)g, (as3_v4i)lds, 0, 0);
}
__device__ __forceinline__ void wait_async_all() {
#if __has_builtin(__builtin_amdgcn_s_wait_asynccnt)
  __builtin_amdgcn_s_wait_asynccnt(0);
#else
  asm volatile("s_wait_asynccnt 0x0" ::: "memory");
#endif
}
#endif

// ------------------------------------------- tiled fp32 -> f16 with transpose
// src: K x N fp32 (row-major), dst: N x K f16 (row-major). K == N == kD here.
__global__ __launch_bounds__(256) void cvt_t_kernel(
    const float* __restrict__ src, half_t* __restrict__ dst)
{
  __shared__ __align__(16) half_t tile[64 * 72];
  const int bk = blockIdx.x * 64;
  const int bn = blockIdx.y * 64;
  const int tid = threadIdx.x;
  {
    int r = tid >> 2;            // k-local 0..63
    int c = (tid & 3) * 16;      // n-local 0,16,32,48
    const float* gp = src + (size_t)(bk + r) * kD + bn + c;
#pragma unroll
    for (int j = 0; j < 16; ++j)
      tile[(c + j) * 72 + r] = (half_t)gp[j];     // transpose into LDS
  }
  __syncthreads();
  {
    int n = tid >> 2;            // n-local 0..63
    int c = (tid & 3) * 16;      // k-local
    half_t* dp = dst + (size_t)(bn + n) * kD + bk + c;
    *(uint4*)dp       = *(const uint4*)&tile[n * 72 + c];
    *(uint4*)(dp + 8) = *(const uint4*)&tile[n * 72 + c + 8];
  }
}

// --------------------------------------------- LayerNorm (+opt modulation+SiLU)
// mode 0: out = f16(LN(x))        mode 1: out = f16(silu(LN(x)*(1+scale)+shift))
__global__ __launch_bounds__(256) void ln_kernel(
    const float* __restrict__ X, const float* __restrict__ gam,
    const float* __restrict__ bet, const float* __restrict__ embOut,
    half_t* __restrict__ outH, int T, int mode)
{
  __shared__ float red[256];
  const int row = blockIdx.x;          // b*T + t
  const int tid = threadIdx.x;
  const float* xr = X + (size_t)row * kD;
  float vals[4];
  float s = 0.f;
#pragma unroll
  for (int i = 0; i < 4; ++i) { vals[i] = xr[tid + i*256]; s += vals[i]; }
  red[tid] = s; __syncthreads();
  for (int o = 128; o > 0; o >>= 1) { if (tid < o) red[tid] += red[tid+o]; __syncthreads(); }
  const float mean = red[0] * (1.f / kD);
  __syncthreads();
  float v = 0.f;
#pragma unroll
  for (int i = 0; i < 4; ++i) { float dlt = vals[i] - mean; v += dlt * dlt; }
  red[tid] = v; __syncthreads();
  for (int o = 128; o > 0; o >>= 1) { if (tid < o) red[tid] += red[tid+o]; __syncthreads(); }
  const float rstd = rsqrtf(red[0] * (1.f / kD) + 1e-5f);
  const int b = row / T;
#pragma unroll
  for (int i = 0; i < 4; ++i) {
    int j = tid + i*256;
    float xn = (vals[i] - mean) * rstd * gam[j] + bet[j];
    if (mode) {
      float sc = embOut[(size_t)b * 2 * kD + j];
      float sh = embOut[(size_t)b * 2 * kD + kD + j];
      xn = xn * (1.f + sc) + sh;
      xn = xn / (1.f + __expf(-xn));      // SiLU
    }
    outH[(size_t)row * kD + j] = (half_t)xn;
  }
}

// ------------------------------------------------- emb path (tiny, scalar GEMM)
__global__ __launch_bounds__(256) void emb_kernel(
    const float* __restrict__ emb, const float* __restrict__ We,
    const float* __restrict__ be, float* __restrict__ out)
{
  int idx = blockIdx.x * 256 + threadIdx.x;  // [0, B*2D)
  int b = idx >> 11;
  int j = idx & 2047;
  const float* er = emb + (size_t)b * kTE;
  float acc = 0.f;
  for (int t = 0; t < kTE; ++t) {
    float e = er[t];
    e = e / (1.f + __expf(-e));              // SiLU
    acc += e * We[(size_t)t * 2 * kD + j];
  }
  out[idx] = acc + be[j];
}

// ------------------------------------------------------- WMMA f16 GEMM (M,N,K)
// A: MxK f16 row-major.  Bt: NxK f16 row-major (operand pre-transposed).
// Double-buffered LDS; async global->LDS when available (one barrier per tile).
// MODE 0: outH[row*N+col] = (acc+bias)*oscale          (f16)
// MODE 1: outH[((b*H+h)*hd+d)*T + m] = acc+bias        (f16, per-head V^T)
// MODE 2: outF[row*N+col] = acc+bias+resid             (f32)
template<int MODE>
__global__ __launch_bounds__(256) void gemm_kernel(
    const half_t* __restrict__ A, const half_t* __restrict__ Bt,
    const float* __restrict__ bias, const float* __restrict__ resid,
    float* __restrict__ outF, half_t* __restrict__ outH,
    int M, int N, int K, float oscale)
{
  constexpr int TSZ = 128 * 48;                   // one 128x32 tile (stride 48)
  __shared__ __align__(16) half_t As [2 * TSZ];
  __shared__ __align__(16) half_t Bts[2 * TSZ];
  const int tid  = threadIdx.x;
  const int lane = tid & 31;
  const int wave = tid >> 5;
  const int bm = blockIdx.x * 128;
  const int bn = blockIdx.y * 128;
  const int wm = (wave >> 1) * 32;   // 0,32,64,96
  const int wn = (wave & 1) * 64;    // 0,64

  v8f acc[2][4] = {};

  const int r2 = tid >> 2;           // 0..63
  const int c2 = (tid & 3) * 8;      // 0,8,16,24
  const half_t* gA0 = A  + (size_t)(bm + r2)      * K + c2;
  const half_t* gA1 = A  + (size_t)(bm + r2 + 64) * K + c2;
  const half_t* gB0 = Bt + (size_t)(bn + r2)      * K + c2;
  const half_t* gB1 = Bt + (size_t)(bn + r2 + 64) * K + c2;
  const int la0 = r2 * 48 + c2;
  const int la1 = (r2 + 64) * 48 + c2;
  const int nk = K >> 5;

  auto compute_tile = [&](const half_t* Asb, const half_t* Bsb) {
    v16h af[2], bf[4];
#pragma unroll
    for (int mi = 0; mi < 2; ++mi) {             // A frag: 2x ds_load_b128
      int row = wm + mi*16 + (lane & 15);
      int kb  = (lane >> 4) * 8;
      V16H t;
      t.u[0] = *(const uint4*)&Asb[row*48 + kb];
      t.u[1] = *(const uint4*)&Asb[row*48 + kb + 16];
      af[mi] = t.v;
    }
#pragma unroll
    for (int ni = 0; ni < 4; ++ni) {             // B frag: 2x ds_load_b128
      int col = wn + ni*16 + (lane & 15);
      int kb  = (lane >> 4) * 8;
      V16H t;
      t.u[0] = *(const uint4*)&Bsb[col*48 + kb];
      t.u[1] = *(const uint4*)&Bsb[col*48 + kb + 16];
      bf[ni] = t.v;
    }
#pragma unroll
    for (int mi = 0; mi < 2; ++mi)
#pragma unroll
      for (int ni = 0; ni < 4; ++ni)
        acc[mi][ni] = __builtin_amdgcn_wmma_f32_16x16x32_f16(
            false, af[mi], false, bf[ni], (short)0, acc[mi][ni], false, false);
  };

#if ASYNC_LDS
  // ---- async pipeline: tiles land in LDS via GLOBAL_LOAD_ASYNC_TO_LDS_B128
  async_cp16(&As [la0], gA0);  async_cp16(&As [la1], gA1);
  async_cp16(&Bts[la0], gB0);  async_cp16(&Bts[la1], gB1);
  wait_async_all();
  __syncthreads();
  for (int t = 0; t < nk; ++t) {
    const int cur = (t & 1) * TSZ;
    const int nxt = ((t + 1) & 1) * TSZ;
    if (t + 1 < nk) {                     // issue next tile before computing
      const int ko = (t + 1) * 32;
      async_cp16(&As [nxt + la0], gA0 + ko);
      async_cp16(&As [nxt + la1], gA1 + ko);
      async_cp16(&Bts[nxt + la0], gB0 + ko);
      async_cp16(&Bts[nxt + la1], gB1 + ko);
    }
    compute_tile(As + cur, Bts + cur);
    wait_async_all();
    __syncthreads();
  }
#else
  // ---- register-staged double buffer (one barrier per tile)
  uint4 ra0 = *(const uint4*)gA0, ra1 = *(const uint4*)gA1;
  uint4 rb0 = *(const uint4*)gB0, rb1 = *(const uint4*)gB1;
  *(uint4*)&As [la0] = ra0;  *(uint4*)&As [la1] = ra1;
  *(uint4*)&Bts[la0] = rb0;  *(uint4*)&Bts[la1] = rb1;
  __syncthreads();
  for (int t = 0; t < nk; ++t) {
    const int cur = (t & 1) * TSZ;
    const int nxt = ((t + 1) & 1) * TSZ;
    if (t + 1 < nk) {
      const int ko = (t + 1) * 32;
      ra0 = *(const uint4*)(gA0 + ko);  ra1 = *(const uint4*)(gA1 + ko);
      rb0 = *(const uint4*)(gB0 + ko);  rb1 = *(const uint4*)(gB1 + ko);
    }
    compute_tile(As + cur, Bts + cur);
    if (t + 1 < nk) {
      *(uint4*)&As [nxt + la0] = ra0;  *(uint4*)&As [nxt + la1] = ra1;
      *(uint4*)&Bts[nxt + la0] = rb0;  *(uint4*)&Bts[nxt + la1] = rb1;
    }
    __syncthreads();
  }
#endif

  const int g = lane >> 4, colL = lane & 15;
#pragma unroll
  for (int mi = 0; mi < 2; ++mi)
#pragma unroll
    for (int ni = 0; ni < 4; ++ni) {
      const int col = bn + wn + ni*16 + colL;
      const float bval = bias[col];
#pragma unroll
      for (int r = 0; r < 8; ++r) {
        const int row = bm + wm + mi*16 + r + 8*g;
        float v = acc[mi][ni][r] + bval;
        if (MODE == 2) {
          v += resid[(size_t)row * N + col];
          outF[(size_t)row * N + col] = v;
        } else if (MODE == 0) {
          outH[(size_t)row * N + col] = (half_t)(v * oscale);
        } else {   // MODE 1: per-head transposed V: [b, h, d, m]
          int bb = row >> 9;           // / kT
          int m  = row & (kT - 1);
          int hh = col >> 6;           // / kHD
          int d  = col & (kHD - 1);
          outH[(((size_t)bb * kH + hh) * kHD + d) * kT + m] = (half_t)v;
        }
      }
    }
}

// ------------------------------------------------------------------- attention
// One block per (b, h, 32-row query tile). Exact softmax over full T=512 keys.
// Q,K: [b, t, h*64+d] f16.  Vt: [b, h, d, m] f16 (pre-transposed by V GEMM).
__global__ __launch_bounds__(256) void attn_kernel(
    const half_t* __restrict__ Q, const half_t* __restrict__ K,
    const half_t* __restrict__ Vt, const float* __restrict__ mask,
    float* __restrict__ Y)
{
  __shared__ __align__(16) half_t Qs[32 * 72];     // 32x64 Q tile
  __shared__ __align__(16) half_t KVs[128 * 72];   // K chunk 128x64 / Vt 64x128
  __shared__ __align__(16) half_t Ps[32 * 520];    // 32x512 scores / probs

  const int tid  = threadIdx.x;
  const int lane = tid & 31;
  const int wave = tid >> 5;
  const int nqb  = kT / 32;                        // 16
  const int qb = blockIdx.x % nqb;
  const int hh = (blockIdx.x / nqb) % kH;
  const int b  = blockIdx.x / (nqb * kH);
  const int q0 = qb * 32;
  const size_t headoff = (size_t)hh * kHD;

  {   // Q tile 32x64
    int r = tid >> 3;                  // 0..31
    int c = (tid & 7) * 8;             // 0..56
    *(uint4*)&Qs[r*72 + c] =
        *(const uint4*)(Q + ((size_t)(b*kT + q0 + r) * kD + headoff + c));
  }

  // ---- phase 1: S = (Q/8) K^T, tiled over 4 chunks of 128 keys
  for (int ch = 0; ch < 4; ++ch) {
    const int m0 = ch * 128;
    {
      int r = tid >> 1;                // 0..127
      int c0 = (tid & 1) * 32;
      const half_t* gp = K + ((size_t)(b*kT + m0 + r) * kD + headoff + c0);
#pragma unroll
      for (int j = 0; j < 4; ++j)
        *(uint4*)&KVs[r*72 + c0 + j*8] = *(const uint4*)(gp + j*8);
      if (ch < 3) __builtin_prefetch(gp + (size_t)128 * kD, 0, 3);
    }
    __syncthreads();
    const int mj = wave;               // each wave: one 16-key strip
#pragma unroll
    for (int mi = 0; mi < 2; ++mi) {
      v8f acc = {};
#pragma unroll
      for (int s = 0; s < 2; ++s) {
        V16H a, bb;
        int arow = mi*16 + (lane & 15);
        int ak   = s*32 + 8*(lane >> 4);
        a.u[0] = *(const uint4*)&Qs[arow*72 + ak];
        a.u[1] = *(const uint4*)&Qs[arow*72 + ak + 16];
        int brow = mj*16 + (lane & 15);          // B[k][n] = K[n][k] contiguous
        int bk   = s*32 + 8*(lane >> 4);
        bb.u[0] = *(const uint4*)&KVs[brow*72 + bk];
        bb.u[1] = *(const uint4*)&KVs[brow*72 + bk + 16];
        acc = __builtin_amdgcn_wmma_f32_16x16x32_f16(
            false, a.v, false, bb.v, (short)0, acc, false, false);
      }
      int g = lane >> 4, col = lane & 15;
#pragma unroll
      for (int r = 0; r < 8; ++r)
        Ps[(mi*16 + r + 8*g)*520 + m0 + mj*16 + col] = (half_t)acc[r];
    }
    __syncthreads();
  }

  // ---- phase 2: masked softmax over 512 keys; 8-thread teams per row
  {
    const int row = tid >> 3;
    const int t8  = tid & 7;
    const float* mrow = mask + ((size_t)b*kT + q0 + row) * kT;
    half_t* prow = Ps + row * 520;
    float mx = -3.0e38f;
    for (int j = t8; j < kT; j += 8) {
      float s = (float)prow[j] + (1.f - mrow[j]) * -100000.f;
      mx = fmaxf(mx, s);
    }
#pragma unroll
    for (int o = 4; o >= 1; o >>= 1) mx = fmaxf(mx, __shfl_xor(mx, o, 8));
    float sum = 0.f;
    for (int j = t8; j < kT; j += 8) {
      float s = (float)prow[j] + (1.f - mrow[j]) * -100000.f;
      float p = __expf(s - mx);
      sum += p;
      prow[j] = (half_t)p;
    }
#pragma unroll
    for (int o = 4; o >= 1; o >>= 1) sum += __shfl_xor(sum, o, 8);
    float inv = 1.f / sum;
    for (int j = t8; j < kT; j += 8) prow[j] = (half_t)((float)prow[j] * inv);
  }
  __syncthreads();

  // ---- phase 3: Y = P @ V, V^T chunks staged so both frags are b128 loads
  const int mi = wave >> 2, di = wave & 3;   // wave -> (qtile, d-strip)
  v8f acc = {};
  const size_t vtbase = ((size_t)b * kH + hh) * kHD * kT;
  for (int ch = 0; ch < 4; ++ch) {
    const int m0 = ch * 128;
    {   // V^T chunk: 64 d-rows x 128 m  (stride 136 halves)
      int r = tid >> 2;                // d 0..63
      int c = (tid & 3) * 32;          // m-local 0,32,64,96
      const half_t* gp = Vt + vtbase + (size_t)r * kT + m0 + c;
#pragma unroll
      for (int j = 0; j < 4; ++j)
        *(uint4*)&KVs[r*136 + c + j*8] = *(const uint4*)(gp + j*8);
      if (ch < 3) __builtin_prefetch(gp + 128, 0, 3);
    }
    __syncthreads();
#pragma unroll
    for (int s = 0; s < 4; ++s) {
      V16H a, bb;
      int arow = mi*16 + (lane & 15);
      int ak   = m0 + s*32 + 8*(lane >> 4);
      a.u[0] = *(const uint4*)&Ps[arow*520 + ak];
      a.u[1] = *(const uint4*)&Ps[arow*520 + ak + 16];
      int dcol = di*16 + (lane & 15);           // B[k=m][n=d] = Vt[d][m]
      int bk2  = s*32 + 8*(lane >> 4);
      bb.u[0] = *(const uint4*)&KVs[dcol*136 + bk2];
      bb.u[1] = *(const uint4*)&KVs[dcol*136 + bk2 + 16];
      acc = __builtin_amdgcn_wmma_f32_16x16x32_f16(
          false, a.v, false, bb.v, (short)0, acc, false, false);
    }
    __syncthreads();
  }
  {
    int g = lane >> 4, col = lane & 15;
#pragma unroll
    for (int r = 0; r < 8; ++r) {
      int rowg = b*kT + q0 + mi*16 + r + 8*g;
      Y[(size_t)rowg * kD + headoff + di*16 + col] = acc[r];
    }
  }
}

// =============================================================== host launcher
extern "C" void kernel_launch(void* const* d_in, const int* in_sizes, int n_in,
                              void* d_out, int out_size, void* d_ws, size_t ws_size,
                              hipStream_t stream) {
  (void)in_sizes; (void)n_in; (void)out_size; (void)ws_size;
  const float* x    = (const float*)d_in[0];
  const float* emb  = (const float*)d_in[1];
  const float* mask = (const float*)d_in[2];
  const float* ln_g = (const float*)d_in[3];
  const float* ln_b = (const float*)d_in[4];
  const float* Wq   = (const float*)d_in[5];
  const float* bq   = (const float*)d_in[6];
  const float* Wk   = (const float*)d_in[7];
  const float* bk   = (const float*)d_in[8];
  const float* Wv   = (const float*)d_in[9];
  const float* bv   = (const float*)d_in[10];
  const float* We   = (const float*)d_in[11];
  const float* be   = (const float*)d_in[12];
  const float* ln2g = (const float*)d_in[13];
  const float* ln2b = (const float*)d_in[14];
  const float* Wo   = (const float*)d_in[15];
  const float* bo   = (const float*)d_in[16];
  float* out = (float*)d_out;

  char* wsb = (char*)d_ws;
  size_t off = 0;
  auto alloc = [&](size_t bytes) -> void* {
    void* p = wsb + off;
    off += (bytes + 255) & ~(size_t)255;
    return p;
  };
  const size_t MT = (size_t)kB * kT;       // 4096 rows
  half_t* xnH = (half_t*)alloc(MT * kD * 2);
  half_t* WqT = (half_t*)alloc((size_t)kD * kD * 2);
  half_t* WkT = (half_t*)alloc((size_t)kD * kD * 2);
  half_t* WvT = (half_t*)alloc((size_t)kD * kD * 2);
  half_t* WoT = (half_t*)alloc((size_t)kD * kD * 2);
  half_t* qH  = (half_t*)alloc(MT * kD * 2);
  half_t* kHh = (half_t*)alloc(MT * kD * 2);
  half_t* vT  = (half_t*)alloc(MT * kD * 2);   // [b,h,d,m]
  float*  yb  = (float*)alloc(MT * kD * 4);
  half_t* h2H = (half_t*)alloc(MT * kD * 2);
  float*  embOut = (float*)alloc((size_t)kB * 2 * kD * 4);

  dim3 tg(kD / 64, kD / 64);               // 16x16 transpose tiles
  cvt_t_kernel<<<tg, 256, 0, stream>>>(Wq, WqT);
  cvt_t_kernel<<<tg, 256, 0, stream>>>(Wk, WkT);
  cvt_t_kernel<<<tg, 256, 0, stream>>>(Wv, WvT);
  cvt_t_kernel<<<tg, 256, 0, stream>>>(Wo, WoT);

  ln_kernel<<<(int)MT, 256, 0, stream>>>(x, ln_g, ln_b, nullptr, xnH, kT, 0);

  dim3 gg((unsigned)(MT / 128), (unsigned)(kD / 128));
  // Q gets the 1/sqrt(64) score scale folded into its f16 store.
  gemm_kernel<0><<<gg, 256, 0, stream>>>(xnH, WqT, bq, nullptr, nullptr, qH,
                                         (int)MT, kD, kD, 0.125f);
  gemm_kernel<0><<<gg, 256, 0, stream>>>(xnH, WkT, bk, nullptr, nullptr, kHh,
                                         (int)MT, kD, kD, 1.0f);
  gemm_kernel<1><<<gg, 256, 0, stream>>>(xnH, WvT, bv, nullptr, nullptr, vT,
                                         (int)MT, kD, kD, 1.0f);

  emb_kernel<<<(kB * 2 * kD) / 256, 256, 0, stream>>>(emb, We, be, embOut);

  attn_kernel<<<kB * kH * (kT / 32), 256, 0, stream>>>(qH, kHh, vT, mask, yb);

  ln_kernel<<<(int)MT, 256, 0, stream>>>(yb, ln2g, ln2b, embOut, h2H, kT, 1);

  gemm_kernel<2><<<gg, 256, 0, stream>>>(h2H, WoT, bo, x, out, nullptr,
                                         (int)MT, kD, kD, 1.0f);
}
